// GAT_53592601920047
// MI455X (gfx1250) — matrix operand order repaired and checked by reading, not compiled, not run on hardware
//
#include <hip/hip_runtime.h>
#include <hip/hip_bf16.h>
#include <math.h>

typedef __attribute__((ext_vector_type(16))) _Float16 v16h;
typedef __attribute__((ext_vector_type(8)))  float    v8f;
typedef unsigned int u32t;
typedef __attribute__((ext_vector_type(4))) u32t v4u;
typedef __attribute__((ext_vector_type(8))) int  v8i;
typedef __attribute__((ext_vector_type(4))) int  v4i;

#define GAT_NEG_SLOPE 0.2f

#if defined(__has_builtin)
#if __has_builtin(__builtin_amdgcn_tensor_load_to_lds) && __has_builtin(__builtin_amdgcn_s_wait_tensorcnt)
#define USE_TDM 1
#endif
#endif
#ifndef USE_TDM
#define USE_TDM 0
#endif

__device__ __forceinline__ void atomicMaxF(float* addr, float val) {
  // sign-aware float max via integer atomics; init value must be -inf
  if (val >= 0.0f) atomicMax((int*)addr, __float_as_int(val));
  else             atomicMin((unsigned int*)addr, __float_as_uint(val));
}

#if USE_TDM
// TDM: DMA a dense 2D f32 tile (height x width, row-major, stride == width)
// from global memory into LDS. Descriptor per CDNA5 ISA ch.8 (D# groups 0/1).
__device__ __forceinline__ void tdm_load_2d_f32(const float* gsrc, void* lds_dst,
                                                int width, int height) {
  u32t lds_off = (u32t)(uintptr_t)lds_dst;           // low 32b of generic = LDS offset
  unsigned long long ga = (unsigned long long)(uintptr_t)gsrc;
  v4u g0;
  g0[0] = 1u;                                        // count=1, user-mode, no gather
  g0[1] = lds_off;                                   // lds_addr
  g0[2] = (u32t)ga;                                  // global_addr[31:0]
  g0[3] = (u32t)(ga >> 32) | (2u << 30);             // global_addr[56:32] | type=2
  v8i g1;
  g1[0] = (int)(2u << 16);                           // wg_mask=0, data_size=4B, no flags
  g1[1] = (int)((u32t)(width & 0xFFFF) << 16);       // tensor_dim0[15:0]
  g1[2] = (int)(((u32t)width >> 16) |                // tensor_dim0[31:16]
                ((u32t)(height & 0xFFFF) << 16));    // tensor_dim1[15:0]
  g1[3] = (int)(((u32t)height >> 16) |               // tensor_dim1[31:16]
                ((u32t)(width & 0xFFFF) << 16));     // tile_dim0
  g1[4] = (int)(height & 0xFFFF);                    // tile_dim1 (tile_dim2=0)
  g1[5] = (int)width;                                // tensor_dim0_stride[31:0]
  g1[6] = 0;                                         // stride hi / dim1_stride lo
  g1[7] = 0;
  v4i z4 = {0, 0, 0, 0};
#if defined(__clang_major__) && (__clang_major__ >= 23)
  v8i z8 = {0, 0, 0, 0, 0, 0, 0, 0};
  __builtin_amdgcn_tensor_load_to_lds(g0, g1, z4, z4, z8, 0);
#else
  __builtin_amdgcn_tensor_load_to_lds(g0, g1, z4, z4, 0);
#endif
}
#endif

__global__ void fill_kernel(float* __restrict__ p, float v, int n) {
  int i = blockIdx.x * blockDim.x + threadIdx.x;
  if (i < n) p[i] = v;
}

// C[M, TN*16] = A[M, KK] @ B[KK, TN*16]; f32 in/out, f16 WMMA, f32 accum.
// Block = 256 threads = 8 waves; each wave owns one 16-row strip and all TN
// column tiles (A fragment reused TN times). B is DMA'd (TDM) into LDS raw,
// then converted/transposed to f16 fragment order so each B fragment is a
// 32-byte contiguous LDS read (2x ds_load_b128). All TN fragments of a K-chunk
// are preloaded before the WMMA burst so the TN WMMAs issue back-to-back.
template <int TN, int KK>
__global__ __launch_bounds__(256) void gemm_wmma_staged(const float* __restrict__ A,
                                                        const float* __restrict__ B,
                                                        float* __restrict__ C,
                                                        int Mtiles) {
  constexpr int NC = TN * 16;
  __shared__ __align__(32) _Float16 sB[NC * KK];     // [n][k], k contiguous
#if USE_TDM
  __shared__ __align__(16) float sF32[KK * NC];      // raw DMA landing zone
  if (threadIdx.x < 32) {                            // one wave issues the DMA
    tdm_load_2d_f32(B, (void*)sF32, NC, KK);
    __builtin_amdgcn_s_wait_tensorcnt(0);
  }
  __syncthreads();
  // convert + transpose LDS(f32 row-major) -> LDS(f16 fragment order)
  for (int idx = threadIdx.x; idx < KK * NC; idx += 256) {
    int k = idx / NC, n = idx % NC;
    sB[n * KK + k] = (_Float16)sF32[idx];
  }
#else
  for (int idx = threadIdx.x; idx < KK * NC; idx += 256) {
    int k = idx / NC, n = idx % NC;
    sB[n * KK + k] = (_Float16)B[idx];
  }
#endif
  __syncthreads();

  const int wave = threadIdx.x >> 5;
  const int lane = threadIdx.x & 31;
  const int tm   = blockIdx.x * 8 + wave;
  if (tm >= Mtiles) return;
  const int hlane = lane & 15;     // row (A) / col (B,D) within tile
  const int hi    = lane >> 4;     // half-wave select

  v8f acc[TN] = {};
  const float* arow = A + (size_t)(tm * 16 + hlane) * KK;
  for (int kc = 0; kc < KK; kc += 32) {
    // A fragment (16x32 f16): lanes 0-15 K 0-15, lanes 16-31 K 16-31
    const float* ap = arow + kc + hi * 16;
    float4 a0 = *(const float4*)(ap);
    float4 a1 = *(const float4*)(ap + 4);
    float4 a2 = *(const float4*)(ap + 8);
    float4 a3 = *(const float4*)(ap + 12);
    v16h af;
    af[0]=(_Float16)a0.x;  af[1]=(_Float16)a0.y;  af[2]=(_Float16)a0.z;  af[3]=(_Float16)a0.w;
    af[4]=(_Float16)a1.x;  af[5]=(_Float16)a1.y;  af[6]=(_Float16)a1.z;  af[7]=(_Float16)a1.w;
    af[8]=(_Float16)a2.x;  af[9]=(_Float16)a2.y;  af[10]=(_Float16)a2.z; af[11]=(_Float16)a2.w;
    af[12]=(_Float16)a3.x; af[13]=(_Float16)a3.y; af[14]=(_Float16)a3.z; af[15]=(_Float16)a3.w;
    // preload all TN B fragments for this K-chunk (issues as a ds_load clause)
    v16h bfr[TN];
#pragma unroll
    for (int t = 0; t < TN; ++t)
      bfr[t] = *(const v16h*)&sB[(t * 16 + hlane) * KK + kc + hi * 16];
    // WMMA burst: TN back-to-back matrix ops on one A fragment
#pragma unroll
    for (int t = 0; t < TN; ++t)
      acc[t] = __builtin_amdgcn_wmma_f32_16x16x32_f16(false, af, false, bfr[t],
                                                      (short)0, acc[t], false, false);
  }
  // D store: lanes 0-15 -> rows +0..7, lanes 16-31 -> rows +8..15
  const int rBase = tm * 16 + hi * 8;
#pragma unroll
  for (int t = 0; t < TN; ++t) {
    float* cp = C + (size_t)rBase * NC + (t * 16 + hlane);
#pragma unroll
    for (int v = 0; v < 8; ++v) cp[(size_t)v * NC] = acc[t][v];
  }
}

// a_s[n,h] = dot(xp[n,h,:], a_src[h,:]); a_d likewise. One thread per (node,head).
__global__ void node_scores(const float* __restrict__ xp,
                            const float* __restrict__ asrc,
                            const float* __restrict__ adst,
                            float* __restrict__ as_, float* __restrict__ ad_,
                            int n, int Hh, int Ch) {
  int i = blockIdx.x * blockDim.x + threadIdx.x;
  if (i >= n * Hh) return;
  int node = i / Hh, h = i % Hh;
  const float* xr = xp + (size_t)node * Hh * Ch + (size_t)h * Ch;
  const float* ar = asrc + h * Ch;
  const float* br = adst + h * Ch;
  float s = 0.f, d = 0.f;
#pragma unroll 8
  for (int c = 0; c < Ch; ++c) { float v = xr[c]; s += v * ar[c]; d += v * br[c]; }
  as_[i] = s;
  ad_[i] = d;
}

__device__ __forceinline__ void edge_sd(const int* ei, int Eg, int e, int& s, int& d) {
  if (e < Eg) { s = ei[e]; d = ei[Eg + e]; }
  else        { s = d = e - Eg; }     // self-loops appended
}

// Pass 1: segment max of leaky-relu logits over destination nodes.
__global__ void edge_max_kernel(const int* __restrict__ ei, int Eg, int ET,
                                const float* __restrict__ as_, const float* __restrict__ ad_,
                                float* __restrict__ mbuf, int Hh) {
  int i = blockIdx.x * blockDim.x + threadIdx.x;
  if (i >= ET * Hh) return;
  int e = i / Hh, h = i % Hh;
  int s, d; edge_sd(ei, Eg, e, s, d);
  float lg = as_[s * Hh + h] + ad_[d * Hh + h];
  lg = lg > 0.f ? lg : GAT_NEG_SLOPE * lg;
  atomicMaxF(&mbuf[d * Hh + h], lg);
}

// Pass 2: denom[dst] += exp(logit - m[dst])
__global__ void edge_denom_kernel(const int* __restrict__ ei, int Eg, int ET,
                                  const float* __restrict__ as_, const float* __restrict__ ad_,
                                  const float* __restrict__ mbuf, float* __restrict__ den,
                                  int Hh) {
  int i = blockIdx.x * blockDim.x + threadIdx.x;
  if (i >= ET * Hh) return;
  int e = i / Hh, h = i % Hh;
  int s, d; edge_sd(ei, Eg, e, s, d);
  float lg = as_[s * Hh + h] + ad_[d * Hh + h];
  lg = lg > 0.f ? lg : GAT_NEG_SLOPE * lg;
  atomicAdd(&den[d * Hh + h], __expf(lg - mbuf[d * Hh + h]));
}

// Pass 3: agg[dst, :] += alpha * xp[src, :]. One wave32 per edge.
__global__ void edge_aggregate_kernel(const int* __restrict__ ei, int Eg, int ET,
                                      const float* __restrict__ xp,
                                      const float* __restrict__ as_, const float* __restrict__ ad_,
                                      const float* __restrict__ mbuf, const float* __restrict__ den,
                                      float* __restrict__ agg, int Hh, int Ch) {
  int gid  = blockIdx.x * blockDim.x + threadIdx.x;
  int e    = gid >> 5;
  int lane = gid & 31;
  if (e >= ET) return;
  int s, d; edge_sd(ei, Eg, e, s, d);
  const int C = Hh * Ch;
  // hide the dependent gather: prefetch this lane's source-feature element
  __builtin_prefetch(xp + (size_t)s * C + lane, 0, 0);
  for (int c = lane; c < C; c += 32) {
    int h = c / Ch;
    float lg = as_[s * Hh + h] + ad_[d * Hh + h];
    lg = lg > 0.f ? lg : GAT_NEG_SLOPE * lg;
    float alpha = __expf(lg - mbuf[d * Hh + h]) / den[d * Hh + h];
    atomicAdd(&agg[(size_t)d * C + c], alpha * xp[(size_t)s * C + c]);
  }
}

__global__ void bias_relu_inplace(float* __restrict__ a, const float* __restrict__ b,
                                  int total, int C) {
  int i = blockIdx.x * blockDim.x + threadIdx.x;
  if (i >= total) return;
  float v = a[i] + b[i % C];
  a[i] = v > 0.f ? v : 0.f;
}

__global__ void bias_out(const float* __restrict__ a, const float* __restrict__ b,
                         float* __restrict__ out, int total, int C) {
  int i = blockIdx.x * blockDim.x + threadIdx.x;
  if (i >= total) return;
  out[i] = a[i] + b[i % C];
}

static inline int grid1(long long n, int blk) { return (int)((n + blk - 1) / blk); }

extern "C" void kernel_launch(void* const* d_in, const int* in_sizes, int n_in,
                              void* d_out, int out_size, void* d_ws, size_t ws_size,
                              hipStream_t stream) {
  (void)n_in; (void)ws_size; (void)out_size;
  const float* x    = (const float*)d_in[0];
  const int*   ei   = (const int*)d_in[1];
  const float* W1   = (const float*)d_in[2];
  const float* asr1 = (const float*)d_in[3];
  const float* ads1 = (const float*)d_in[4];
  const float* b1   = (const float*)d_in[5];
  const float* W2   = (const float*)d_in[6];
  const float* asr2 = (const float*)d_in[7];
  const float* ads2 = (const float*)d_in[8];
  const float* b2   = (const float*)d_in[9];
  float* out = (float*)d_out;

  const int DIN = 128, H1 = 4, C1 = 32, DOUT = 32;
  const int C1t = H1 * C1;                    // 128
  const int Nn  = in_sizes[0] / DIN;          // 50000
  const int Eg  = in_sizes[1] / 2;            // 800000
  const int ET  = Eg + Nn;                    // + self loops
  const int Mt  = Nn / 16;                    // 3125 row tiles (exact)

  float* ws = (float*)d_ws;
  size_t o = 0;
  float* xp1  = ws + o; o += (size_t)Nn * C1t;
  float* agg1 = ws + o; o += (size_t)Nn * C1t;  // becomes h after relu-in-place
  float* as1  = ws + o; o += (size_t)Nn * H1;
  float* ad1  = ws + o; o += (size_t)Nn * H1;
  float* m1   = ws + o; o += (size_t)Nn * H1;
  float* dn1  = ws + o; o += (size_t)Nn * H1;
  float* xp2  = ws + o; o += (size_t)Nn * DOUT;
  float* agg2 = ws + o; o += (size_t)Nn * DOUT;
  float* as2  = ws + o; o += (size_t)Nn;
  float* ad2  = ws + o; o += (size_t)Nn;
  float* m2   = ws + o; o += (size_t)Nn;
  float* dn2  = ws + o; o += (size_t)Nn;

  const int BLK = 256;
  const float NEG_INF = -INFINITY;

  // ---- init accumulators (graph-replay safe: done every call)
  fill_kernel<<<grid1((long long)Nn * C1t, BLK), BLK, 0, stream>>>(agg1, 0.f, Nn * C1t);
  fill_kernel<<<grid1((long long)Nn * H1, BLK), BLK, 0, stream>>>(dn1, 0.f, Nn * H1);
  fill_kernel<<<grid1((long long)Nn * H1, BLK), BLK, 0, stream>>>(m1, NEG_INF, Nn * H1);
  fill_kernel<<<grid1((long long)Nn * DOUT, BLK), BLK, 0, stream>>>(agg2, 0.f, Nn * DOUT);
  fill_kernel<<<grid1(Nn, BLK), BLK, 0, stream>>>(dn2, 0.f, Nn);
  fill_kernel<<<grid1(Nn, BLK), BLK, 0, stream>>>(m2, NEG_INF, Nn);

  // ---- layer 1: xp1 = x @ W1  (WMMA, W1 DMA'd to LDS via TDM)
  gemm_wmma_staged<8, 128><<<grid1(Mt, 8), BLK, 0, stream>>>(x, W1, xp1, Mt);
  node_scores<<<grid1((long long)Nn * H1, BLK), BLK, 0, stream>>>(xp1, asr1, ads1, as1, ad1, Nn, H1, C1);
  edge_max_kernel<<<grid1((long long)ET * H1, BLK), BLK, 0, stream>>>(ei, Eg, ET, as1, ad1, m1, H1);
  edge_denom_kernel<<<grid1((long long)ET * H1, BLK), BLK, 0, stream>>>(ei, Eg, ET, as1, ad1, m1, dn1, H1);
  edge_aggregate_kernel<<<grid1((long long)ET * 32, BLK), BLK, 0, stream>>>(
      ei, Eg, ET, xp1, as1, ad1, m1, dn1, agg1, H1, C1);
  bias_relu_inplace<<<grid1((long long)Nn * C1t, BLK), BLK, 0, stream>>>(agg1, b1, Nn * C1t, C1t);

  // ---- layer 2: xp2 = h @ W2  (WMMA, W2 DMA'd to LDS via TDM)
  gemm_wmma_staged<2, 128><<<grid1(Mt, 8), BLK, 0, stream>>>(agg1, W2, xp2, Mt);
  node_scores<<<grid1(Nn, BLK), BLK, 0, stream>>>(xp2, asr2, ads2, as2, ad2, Nn, 1, DOUT);
  edge_max_kernel<<<grid1(ET, BLK), BLK, 0, stream>>>(ei, Eg, ET, as2, ad2, m2, 1);
  edge_denom_kernel<<<grid1(ET, BLK), BLK, 0, stream>>>(ei, Eg, ET, as2, ad2, m2, dn2, 1);
  edge_aggregate_kernel<<<grid1((long long)ET * 32, BLK), BLK, 0, stream>>>(
      ei, Eg, ET, xp2, as2, ad2, m2, dn2, agg2, 1, DOUT);
  bias_out<<<grid1((long long)Nn * DOUT, BLK), BLK, 0, stream>>>(agg2, b2, out, Nn * DOUT, DOUT);
}
